// TinyRecursiveReasoningModel_ACTV1Block_33071248179636
// MI455X (gfx1250) — matrix operand order repaired and checked by reading, not compiled, not run on hardware
//
#include <hip/hip_runtime.h>
#include <hip/hip_bf16.h>

// ---------------- problem constants ----------------
constexpr int cB  = 4;
constexpr int cL  = 2048;
constexpr int cDM = 1024;
constexpr int cE  = 2048;     // EXPAND * DM
constexpr int cN  = 64;
constexpr int cR  = 64;       // DM/16
constexpr int cK  = 4;
constexpr int cP  = cR + 2 * cN;   // 192
constexpr int cI  = 1536;          // INTER
constexpr float cEPS = 1e-5f;

typedef __attribute__((ext_vector_type(16))) __bf16 bf16x16;
typedef __attribute__((ext_vector_type(8)))  __bf16 bf16x8;
typedef __attribute__((ext_vector_type(8)))  float  f32x8;

// =====================================================================
// fp32 -> bf16 bulk convert (n must be a multiple of 4)
// =====================================================================
__global__ void cvt_bf16_kernel(const float* __restrict__ src,
                                __bf16* __restrict__ dst, long long n)
{
    long long i = ((long long)blockIdx.x * blockDim.x + threadIdx.x) * 4;
    if (i >= n) return;
    const float4 v = *(const float4*)(src + i);
    dst[i + 0] = (__bf16)v.x;
    dst[i + 1] = (__bf16)v.y;
    dst[i + 2] = (__bf16)v.z;
    dst[i + 3] = (__bf16)v.w;
}

// =====================================================================
// WMMA GEMM, bf16 operands, fp32 accumulate:
//   D[m,n] = epi( sum_k A[m,k] * B[k,n] )
// A: bf16 row-major, (m,k) at A + m*lda + k          (K contiguous)
// B: bf16 K-major,  (k,n) at B + n*ldb + k           (K contiguous)
// D: fp32 at D + m*sdm + n*sdn (+ blockIdx.z*batchD); optional bf16 copy.
//
// Block tile 64(M) x 128(N), K-step 32.  8 waves arranged 2(M) x 4(N);
// each wave owns a 32x32 sub-tile = 2x2 accumulators = 4 WMMAs per
// K-step from 2 A-fragments and 2 B-fragments (2x register reuse).
// Double-buffered LDS (2 x 12 KB): the async copy of K-step i+1
// (global_load_async_to_lds_b128, ASYNCcnt) is issued before computing
// K-step i; one s_wait_asynccnt + one barrier per iteration.
// EPI: 0 = store, 1 = softplus(x + bias[m])
// =====================================================================
template<int EPI>
__global__ void gemm_bf16_wmma_kernel(
    const __bf16* __restrict__ A, const __bf16* __restrict__ Bm,
    float* __restrict__ D, __bf16* __restrict__ D16,
    const float* __restrict__ bias,
    int M, int N, int Kdim, int lda, int ldb,
    long long sdm, long long sdn,
    long long batchB, long long batchD)
{
    __shared__ __align__(16) __bf16 sA[2][64 * 32];    // [buf][row][k], 4 KB each
    __shared__ __align__(16) __bf16 sB[2][128 * 32];   // [buf][col][k], 8 KB each

    const int tid  = threadIdx.x;
    const int lane = tid & 31;                 // wave32
    const int wave = tid >> 5;
    const int wm   = wave & 1;                 // 2 M-subtiles (32 rows each)
    const int wn   = wave >> 1;                // 4 N-subtiles (32 cols each)
    const int mblk = blockIdx.y * 64;
    const int nblk = blockIdx.x * 128;

    const __bf16* Bp = Bm + (long long)blockIdx.z * batchB;
    float*        Dp = D  + (long long)blockIdx.z * batchD;

    const int r  = lane & 15;                  // A row / B col / D col
    const int hi = lane >> 4;                  // K-half selector

    // per-thread copy coordinates: A = 256 chunks of 16B, B = 512 chunks
    const int aRowC = tid >> 2;                // 0..63
    const int chkC  = tid & 3;                 // 0..3 (8 bf16 each)
    const int bCol0 = tid >> 2;                // 0..63
    const int bCol1 = (tid + 256) >> 2;        // 64..127

    auto stage = [&](int kt, int buf) {
        const __bf16* gA = A + (long long)(mblk + aRowC) * lda + kt + chkC * 8;
        unsigned la = (unsigned)(size_t)(&sA[buf][aRowC * 32 + chkC * 8]);
        asm volatile("global_load_async_to_lds_b128 %0, %1, off"
                     :: "v"(la), "v"(gA) : "memory");
        const __bf16* gB0 = Bp + (long long)(nblk + bCol0) * ldb + kt + chkC * 8;
        unsigned lb0 = (unsigned)(size_t)(&sB[buf][bCol0 * 32 + chkC * 8]);
        asm volatile("global_load_async_to_lds_b128 %0, %1, off"
                     :: "v"(lb0), "v"(gB0) : "memory");
        const __bf16* gB1 = Bp + (long long)(nblk + bCol1) * ldb + kt + chkC * 8;
        unsigned lb1 = (unsigned)(size_t)(&sB[buf][bCol1 * 32 + chkC * 8]);
        asm volatile("global_load_async_to_lds_b128 %0, %1, off"
                     :: "v"(lb1), "v"(gB1) : "memory");
    };

    f32x8 acc00 = {}, acc01 = {}, acc10 = {}, acc11 = {};

    // prologue: fill buffer 0
    stage(0, 0);
    asm volatile("s_wait_asynccnt 0x0" ::: "memory");
    __syncthreads();

    int buf = 0;
    for (int kt = 0; kt < Kdim; kt += 32) {
        // overlap: start DMA of next K-slab into the other buffer
        if (kt + 32 < Kdim) stage(kt + 32, buf ^ 1);

        // ---- fragments from LDS (contiguous k order, same for A and B) ----
        // A 16x32: lanes 0-15 hold K=[0..15], lanes 16-31 K=[16..31]
        const __bf16* a0p = &sA[buf][(wm * 32 + r) * 32];
        const __bf16* a1p = &sA[buf][(wm * 32 + 16 + r) * 32];
        const bf16x8 a0lo = *(const bf16x8*)(a0p + hi * 16);
        const bf16x8 a0hi = *(const bf16x8*)(a0p + hi * 16 + 8);
        const bf16x8 a1lo = *(const bf16x8*)(a1p + hi * 16);
        const bf16x8 a1hi = *(const bf16x8*)(a1p + hi * 16 + 8);
        const bf16x16 af0 = __builtin_shufflevector(
            a0lo, a0hi, 0, 1, 2, 3, 4, 5, 6, 7, 8, 9, 10, 11, 12, 13, 14, 15);
        const bf16x16 af1 = __builtin_shufflevector(
            a1lo, a1hi, 0, 1, 2, 3, 4, 5, 6, 7, 8, 9, 10, 11, 12, 13, 14, 15);
        // B 32x16: lanes 0-15 hold K=[0..15], lanes 16-31 K=[16..31]
        const bf16x16 bf0 = *(const bf16x16*)(&sB[buf][(wn * 32 + r) * 32 + hi * 16]);
        const bf16x16 bf1 = *(const bf16x16*)(&sB[buf][(wn * 32 + 16 + r) * 32 + hi * 16]);

        acc00 = __builtin_amdgcn_wmma_f32_16x16x32_bf16(false, af0, false, bf0, (short)0, acc00, false, false);
        acc01 = __builtin_amdgcn_wmma_f32_16x16x32_bf16(false, af0, false, bf1, (short)0, acc01, false, false);
        acc10 = __builtin_amdgcn_wmma_f32_16x16x32_bf16(false, af1, false, bf0, (short)0, acc10, false, false);
        acc11 = __builtin_amdgcn_wmma_f32_16x16x32_bf16(false, af1, false, bf1, (short)0, acc11, false, false);

        // ensure our next-slab DMA landed, then publish buffer swap
        asm volatile("s_wait_asynccnt 0x0" ::: "memory");
        __syncthreads();
        buf ^= 1;
    }

    // D layout: VGPR rr -> M = rowBase + rr + 8*hi ; col N = colBase + r
    __bf16* D16p = D16 ? (D16 + (long long)blockIdx.z * batchD) : nullptr;
    const f32x8* accs[2][2] = {{&acc00, &acc01}, {&acc10, &acc11}};
    #pragma unroll
    for (int i = 0; i < 2; ++i) {
        #pragma unroll
        for (int j = 0; j < 2; ++j) {
            const int mBase = mblk + wm * 32 + i * 16;
            const int n = nblk + wn * 32 + j * 16 + r;
            const f32x8& a = *accs[i][j];
            #pragma unroll
            for (int rr = 0; rr < 8; ++rr) {
                const int m = mBase + rr + 8 * hi;
                float v = a[rr];
                if (EPI == 1) {
                    v += bias[m];
                    v = (v > 20.f) ? v : __logf(1.f + __expf(v));   // softplus
                }
                const long long di = (long long)m * sdm + (long long)n * sdn;
                Dp[di] = v;
                if (D16p) D16p[di] = (__bf16)v;
            }
        }
    }
}

// =====================================================================
// Causal depthwise conv (K=4) + SiLU, channel-last layouts.
// x = xz_t[:, :, 0:E];  reverse=1 reads time-reversed (backward dir).
// Writes u fp32 and bf16 (both (B,L,E)).
// =====================================================================
__global__ void conv_silu_kernel(const float* __restrict__ xz_t,
                                 const float* __restrict__ w,
                                 const float* __restrict__ bias,
                                 float* __restrict__ u,
                                 __bf16* __restrict__ u16, int reverse)
{
    const long long total = (long long)cB * cL * cE;
    long long idx = (long long)blockIdx.x * blockDim.x + threadIdx.x;
    if (idx >= total) return;
    const int e = (int)(idx % cE);
    long long t = idx / cE;
    const int l = (int)(t % cL);
    const int b = (int)(t / cL);

    float acc = bias[e];
    #pragma unroll
    for (int k = 0; k < cK; ++k) {
        const int j = l - (cK - 1) + k;
        if (j >= 0) {
            const int p = reverse ? (cL - 1 - j) : j;
            acc += w[e * cK + k] * xz_t[((long long)b * cL + p) * (2 * cE) + e];
        }
    }
    const float s = acc / (1.f + __expf(-acc));   // silu
    const long long o = ((long long)b * cL + l) * cE + e;
    u[o] = s;
    u16[o] = (__bf16)s;
}

// =====================================================================
// Selective scan, channel-last.  Block = 64 threads = 64 channels of one
// batch.  h[64], A-row in registers; B_t/C_t staged in LDS per step.
// y = (scan + D*u) * silu(z);  reverse_z mirrors z for the bwd pass.
// =====================================================================
__global__ void scan_kernel(const float* __restrict__ u,       // (B,L,E)
                            const float* __restrict__ delta,   // (B,L,E)
                            const float* __restrict__ xdbl,    // (B,L,192)
                            const float* __restrict__ xz_t,    // (B,L,2E)
                            const float* __restrict__ A_log,   // (E,N)
                            const float* __restrict__ Dp,      // (E)
                            float* __restrict__ y,             // (B,L,E)
                            int reverse_z)
{
    const int b = blockIdx.y;
    const int e = blockIdx.x * 64 + threadIdx.x;

    __shared__ float sB[cN];
    __shared__ float sC[cN];

    float h[cN];
    float Arow[cN];
    #pragma unroll
    for (int n = 0; n < cN; ++n) {
        h[n] = 0.f;
        Arow[n] = -__expf(A_log[(long long)e * cN + n]);
    }
    const float Dv = Dp[e];

    const float* ub = u     + (long long)b * cL * cE + e;
    const float* db = delta + (long long)b * cL * cE + e;
    const float* zb = xz_t  + (long long)b * cL * (2 * cE) + cE + e;
    float*       yb = y     + (long long)b * cL * cE + e;
    const float* xb = xdbl  + (long long)b * cL * cP;

    for (int t = 0; t < cL; ++t) {
        __syncthreads();   // protect previous step's sB/sC
        sB[threadIdx.x] = xb[(long long)t * cP + cR + threadIdx.x];
        sC[threadIdx.x] = xb[(long long)t * cP + cR + cN + threadIdx.x];
        __syncthreads();

        const float dt = db[(long long)t * cE];
        const float ut = ub[(long long)t * cE];
        const float du = dt * ut;
        float acc = 0.f;
        #pragma unroll
        for (int n = 0; n < cN; ++n) {
            h[n] = h[n] * __expf(dt * Arow[n]) + du * sB[n];
            acc += h[n] * sC[n];
        }
        float yv = acc + Dv * ut;
        const int zl = reverse_z ? (cL - 1 - t) : t;
        const float zv = zb[(long long)zl * (2 * cE)];
        yv *= zv / (1.f + __expf(-zv));   // * silu(z)
        yb[(long long)t * cE] = yv;
    }
}

// ycomb16[b,l,e] = bf16( 0.5*(y_f[b,l,e] + y_b_rev[b,L-1-l,e]) )
__global__ void combine_kernel(const float* __restrict__ yf,
                               const float* __restrict__ ybr,
                               __bf16* __restrict__ y16)
{
    const long long total = (long long)cB * cL * cE;
    long long idx = (long long)blockIdx.x * blockDim.x + threadIdx.x;
    if (idx >= total) return;
    const int e = (int)(idx % cE);
    long long t = idx / cE;
    const int l = (int)(t % cL);
    const int b = (int)(t / cL);
    const float v = 0.5f * (yf[idx] +
                            ybr[((long long)b * cL + (cL - 1 - l)) * cE + e]);
    y16[idx] = (__bf16)v;
}

// out = rmsnorm(x1 + x2); optional bf16 copy.  Block = 256 threads/row.
__global__ void rmsnorm_add_kernel(const float* __restrict__ x1,
                                   const float* __restrict__ x2,
                                   float* __restrict__ out,
                                   __bf16* __restrict__ out16)
{
    const long long row = blockIdx.x;   // 0 .. B*L-1
    const float* a = x1 + row * cDM;
    const float* b = x2 + row * cDM;
    __shared__ float red[256];

    float v[4];
    float s = 0.f;
    #pragma unroll
    for (int i = 0; i < 4; ++i) {
        v[i] = a[threadIdx.x + i * 256] + b[threadIdx.x + i * 256];
        s += v[i] * v[i];
    }
    red[threadIdx.x] = s;
    __syncthreads();
    for (int off = 128; off; off >>= 1) {
        if (threadIdx.x < off) red[threadIdx.x] += red[threadIdx.x + off];
        __syncthreads();
    }
    const float inv = rsqrtf(red[0] / (float)cDM + cEPS);
    float* o = out + row * cDM;
    #pragma unroll
    for (int i = 0; i < 4; ++i) {
        const float r = v[i] * inv;
        o[threadIdx.x + i * 256] = r;
        if (out16) out16[row * cDM + threadIdx.x + i * 256] = (__bf16)r;
    }
}

// act16[row,i] = bf16( silu(gu[row,i]) * gu[row, INTER+i] )
__global__ void swiglu_kernel(const float* __restrict__ gu,
                              __bf16* __restrict__ act16)
{
    const long long total = (long long)cB * cL * cI;
    long long idx = (long long)blockIdx.x * blockDim.x + threadIdx.x;
    if (idx >= total) return;
    const long long row = idx / cI;
    const int i = (int)(idx % cI);
    const float g = gu[row * (2 * cI) + i];
    const float uu = gu[row * (2 * cI) + cI + i];
    act16[idx] = (__bf16)((g / (1.f + __expf(-g))) * uu);
}

// ---------------------------------------------------------------------
extern "C" void kernel_launch(void* const* d_in, const int* in_sizes, int n_in,
                              void* d_out, int out_size, void* d_ws, size_t ws_size,
                              hipStream_t stream)
{
    const float* hidden     = (const float*)d_in[0];
    const float* in_proj_w  = (const float*)d_in[1];
    const float* conv_w_f   = (const float*)d_in[2];
    const float* conv_b_f   = (const float*)d_in[3];
    const float* conv_w_b   = (const float*)d_in[4];
    const float* conv_b_b   = (const float*)d_in[5];
    const float* x_proj_w_f = (const float*)d_in[6];
    const float* dt_w_f     = (const float*)d_in[7];
    const float* dt_b_f     = (const float*)d_in[8];
    const float* A_log_f    = (const float*)d_in[9];
    const float* D_f        = (const float*)d_in[10];
    const float* x_proj_w_b = (const float*)d_in[11];
    const float* dt_w_b     = (const float*)d_in[12];
    const float* dt_b_b     = (const float*)d_in[13];
    const float* A_log_b    = (const float*)d_in[14];
    const float* D_b        = (const float*)d_in[15];
    const float* out_proj_w = (const float*)d_in[16];
    const float* gate_up_w  = (const float*)d_in[17];
    const float* down_w     = (const float*)d_in[18];
    float* out = (float*)d_out;
    (void)in_sizes; (void)n_in; (void)out_size; (void)ws_size;

    // -------- workspace carve-up (256B aligned) --------
    char* base = (char*)d_ws;
    size_t off = 0;
    auto carveF = [&](size_t n) {
        off = (off + 255) & ~(size_t)255;
        float* p = (float*)(base + off); off += n * sizeof(float); return p;
    };
    auto carveH = [&](size_t n) {
        off = (off + 255) & ~(size_t)255;
        __bf16* p = (__bf16*)(base + off); off += n * sizeof(__bf16); return p;
    };

    // fp32 intermediates (channel-last layouts)
    float* xz_t    = carveF((size_t)cB * cL * 2 * cE);  // (B,L,2E)
    float* u_f     = carveF((size_t)cB * cL * cE);
    float* u_b     = carveF((size_t)cB * cL * cE);
    float* xdbl_f  = carveF((size_t)cB * cL * cP);
    float* xdbl_b  = carveF((size_t)cB * cL * cP);
    float* delta_f = carveF((size_t)cB * cL * cE);
    float* delta_b = carveF((size_t)cB * cL * cE);
    float* y_f     = carveF((size_t)cB * cL * cE);
    float* y_b     = carveF((size_t)cB * cL * cE);
    float* mamba_o = carveF((size_t)cB * cL * cDM);
    float* hbuf    = carveF((size_t)cB * cL * cDM);
    float* gu      = carveF((size_t)cB * cL * 2 * cI);
    float* mlp     = carveF((size_t)cB * cL * cDM);
    // bf16 operands
    __bf16* hid16  = carveH((size_t)cB * cL * cDM);
    __bf16* wIn16  = carveH((size_t)2 * cE * cDM);
    __bf16* wXf16  = carveH((size_t)cP * cE);
    __bf16* wXb16  = carveH((size_t)cP * cE);
    __bf16* wDf16  = carveH((size_t)cE * cR);
    __bf16* wDb16  = carveH((size_t)cE * cR);
    __bf16* wOut16 = carveH((size_t)cDM * cE);
    __bf16* wGu16  = carveH((size_t)2 * cI * cDM);
    __bf16* wDn16  = carveH((size_t)cDM * cI);
    __bf16* uf16   = carveH((size_t)cB * cL * cE);
    __bf16* ub16   = carveH((size_t)cB * cL * cE);
    __bf16* xdf16  = carveH((size_t)cB * cL * cP);
    __bf16* xdb16  = carveH((size_t)cB * cL * cP);
    __bf16* yc16   = carveH((size_t)cB * cL * cE);
    __bf16* h16    = carveH((size_t)cB * cL * cDM);
    __bf16* act16  = carveH((size_t)cB * cL * cI);

    const dim3 blk(256);
    auto cvt = [&](const float* s, __bf16* d, long long n) {
        dim3 g((unsigned)((n / 4 + 255) / 256));
        cvt_bf16_kernel<<<g, blk, 0, stream>>>(s, d, n);
    };

    // 0) one-time bf16 conversions of weights + hidden
    cvt(hidden,     hid16,  (long long)cB * cL * cDM);
    cvt(in_proj_w,  wIn16,  (long long)2 * cE * cDM);
    cvt(x_proj_w_f, wXf16,  (long long)cP * cE);
    cvt(x_proj_w_b, wXb16,  (long long)cP * cE);
    cvt(dt_w_f,     wDf16,  (long long)cE * cR);
    cvt(dt_w_b,     wDb16,  (long long)cE * cR);
    cvt(out_proj_w, wOut16, (long long)cDM * cE);
    cvt(gate_up_w,  wGu16,  (long long)2 * cI * cDM);
    cvt(down_w,     wDn16,  (long long)cDM * cI);

    // 1) xz_t = in_proj_w (2E x DM) x hidden^T   -> (B, L, 2E)
    {
        dim3 grid(cL / 128, (2 * cE) / 64, cB);
        gemm_bf16_wmma_kernel<0><<<grid, blk, 0, stream>>>(
            wIn16, hid16, xz_t, nullptr, nullptr,
            2 * cE, cL, cDM, /*lda*/ cDM, /*ldb*/ cDM,
            /*sdm*/ 1, /*sdn*/ 2 * cE,
            /*batchB*/ (long long)cL * cDM, /*batchD*/ (long long)cL * 2 * cE);
    }

    // 2) depthwise conv + silu (fwd, and bwd on reversed time)
    {
        long long total = (long long)cB * cL * cE;
        dim3 grid((unsigned)((total + 255) / 256));
        conv_silu_kernel<<<grid, blk, 0, stream>>>(xz_t, conv_w_f, conv_b_f, u_f, uf16, 0);
        conv_silu_kernel<<<grid, blk, 0, stream>>>(xz_t, conv_w_b, conv_b_b, u_b, ub16, 1);
    }

    // 3) x_dbl = x_proj_w (192 x E) x u^T -> (B, L, 192), fp32 + bf16
    {
        dim3 grid(cL / 128, cP / 64, cB);
        gemm_bf16_wmma_kernel<0><<<grid, blk, 0, stream>>>(
            wXf16, uf16, xdbl_f, xdf16, nullptr,
            cP, cL, cE, cE, cE,
            1, cP, (long long)cL * cE, (long long)cL * cP);
        gemm_bf16_wmma_kernel<0><<<grid, blk, 0, stream>>>(
            wXb16, ub16, xdbl_b, xdb16, nullptr,
            cP, cL, cE, cE, cE,
            1, cP, (long long)cL * cE, (long long)cL * cP);
    }

    // 4) delta = softplus(dt_w (E x R) x dtr^T + dt_b) -> (B, L, E)
    {
        dim3 grid(cL / 128, cE / 64, cB);
        gemm_bf16_wmma_kernel<1><<<grid, blk, 0, stream>>>(
            wDf16, xdf16, delta_f, nullptr, dt_b_f,
            cE, cL, cR, cR, cP,
            1, cE, (long long)cL * cP, (long long)cL * cE);
        gemm_bf16_wmma_kernel<1><<<grid, blk, 0, stream>>>(
            wDb16, xdb16, delta_b, nullptr, dt_b_b,
            cE, cL, cR, cR, cP,
            1, cE, (long long)cL * cP, (long long)cL * cE);
    }

    // 5) selective scans (fwd + bwd), gated by silu(z)
    {
        dim3 grid(cE / 64, cB);
        dim3 sblk(64);
        scan_kernel<<<grid, sblk, 0, stream>>>(u_f, delta_f, xdbl_f, xz_t,
                                               A_log_f, D_f, y_f, 0);
        scan_kernel<<<grid, sblk, 0, stream>>>(u_b, delta_b, xdbl_b, xz_t,
                                               A_log_b, D_b, y_b, 1);
    }

    // 6) ycomb16 = bf16( 0.5 * (y_f + reverse(y_b)) )
    {
        long long total = (long long)cB * cL * cE;
        dim3 grid((unsigned)((total + 255) / 256));
        combine_kernel<<<grid, blk, 0, stream>>>(y_f, y_b, yc16);
    }

    // 7) mamba_out = out_proj_w (DM x E) x y^T -> (B, L, DM)
    {
        dim3 grid(cL / 128, cDM / 64, cB);
        gemm_bf16_wmma_kernel<0><<<grid, blk, 0, stream>>>(
            wOut16, yc16, mamba_o, nullptr, nullptr,
            cDM, cL, cE, cE, cE,
            1, cDM, (long long)cL * cE, (long long)cL * cDM);
    }

    // 8) h = rmsnorm(hidden + mamba_out), fp32 + bf16
    {
        dim3 grid(cB * cL);
        rmsnorm_add_kernel<<<grid, blk, 0, stream>>>(hidden, mamba_o, hbuf, h16);
    }

    // 9) gu = gate_up_w (2I x DM) x h^T   (N = B*L)
    {
        dim3 grid((cB * cL) / 128, (2 * cI) / 64, 1);
        gemm_bf16_wmma_kernel<0><<<grid, blk, 0, stream>>>(
            wGu16, h16, gu, nullptr, nullptr,
            2 * cI, cB * cL, cDM, cDM, cDM,
            1, 2 * cI, 0, 0);
    }

    // 10) act16 = bf16( silu(gate) * up )
    {
        long long total = (long long)cB * cL * cI;
        dim3 grid((unsigned)((total + 255) / 256));
        swiglu_kernel<<<grid, blk, 0, stream>>>(gu, act16);
    }

    // 11) mlp = down_w (DM x I) x act^T
    {
        dim3 grid((cB * cL) / 128, cDM / 64, 1);
        gemm_bf16_wmma_kernel<0><<<grid, blk, 0, stream>>>(
            wDn16, act16, mlp, nullptr, nullptr,
            cDM, cB * cL, cI, cI, cI,
            1, cDM, 0, 0);
    }

    // 12) out = rmsnorm(h + mlp)
    {
        dim3 grid(cB * cL);
        rmsnorm_add_kernel<<<grid, blk, 0, stream>>>(hbuf, mlp, out, nullptr);
    }
}
